// StabilityLossCoP_81990925680654
// MI455X (gfx1250) — matrix operand order repaired and checked by reading, not compiled, not run on hardware
//
#include <hip/hip_runtime.h>
#include <hip/hip_bf16.h>
#include <math.h>
#include <stdint.h>

#define NVERT 6890
#define NFACE 13776
#define NHDV  20000
#define NPART 24
#define BATCH 64
#define NCOL  192            // 3*BATCH, col = 3*b + d
#define KPAD  6912           // NVERT padded up to multiple of 64, zero-filled
#define KSTEP 32
#define MBLK  128
#define KFULL (NVERT / KSTEP)   // 215 full steps, tail of 10

typedef __attribute__((ext_vector_type(16))) _Float16 v16h;
typedef __attribute__((ext_vector_type(4)))  _Float16 v4h;
typedef __attribute__((ext_vector_type(8)))  float    v8f;

// ---------------------------------------------------------------------------
// Kernel 1: build f16 B^T operand  bft[col][k] = vertices[b][k][d], col = 3b+d
// Zero-padded to KPAD so GEMM B-side needs no K-tail guard.
// ---------------------------------------------------------------------------
__global__ __launch_bounds__(256) void prep_kernel(
    const float* __restrict__ vertices, _Float16* __restrict__ bft) {
  int idx = blockIdx.x * 256 + threadIdx.x;
  if (idx >= NCOL * KPAD) return;
  int col = idx / KPAD;
  int k   = idx - col * KPAD;
  int b = col / 3;
  int d = col - 3 * b;
  float v = 0.0f;
  if (k < NVERT) v = vertices[(size_t)b * NVERT * 3 + (size_t)k * 3 + d];
  bft[idx] = (_Float16)v;
}

// ---------------------------------------------------------------------------
// Kernel 2: per-part face volumes, one block per batch, deterministic.
// ---------------------------------------------------------------------------
__global__ __launch_bounds__(256) void vol_kernel(
    const float* __restrict__ vertices, const int* __restrict__ faces,
    const int* __restrict__ face_part, float* __restrict__ per_part) {
  __shared__ float acc[NPART][256];
  int tid = threadIdx.x;
  int b = blockIdx.x;
  for (int p = 0; p < NPART; ++p) acc[p][tid] = 0.0f;
  const float* vb = vertices + (size_t)b * NVERT * 3;
  for (int f = tid; f < NFACE; f += 256) {
    int i0 = faces[3 * f + 0], i1 = faces[3 * f + 1], i2 = faces[3 * f + 2];
    float ax = vb[3 * i0], ay = vb[3 * i0 + 1], az = vb[3 * i0 + 2];
    float bx = vb[3 * i1], by = vb[3 * i1 + 1], bz = vb[3 * i1 + 2];
    float cx = vb[3 * i2], cy = vb[3 * i2 + 1], cz = vb[3 * i2 + 2];
    float crx = by * cz - bz * cy;
    float cry = bz * cx - bx * cz;
    float crz = bx * cy - by * cx;
    float vol = fabsf(ax * crx + ay * cry + az * crz) * (1.0f / 6.0f);
    acc[face_part[f]][tid] += vol;
  }
  __syncthreads();
  for (int s = 128; s > 0; s >>= 1) {
    if (tid < s) {
      for (int p = 0; p < NPART; ++p) acc[p][tid] += acc[p][tid + s];
    }
    __syncthreads();
  }
  if (tid < NPART) per_part[b * NPART + tid] = acc[tid][0];
}

// ---------------------------------------------------------------------------
// Kernel 3: GEMM  C[col][n] = sum_k A[n][k] * Bt[col][k]
// Double-buffered LDS; A staged f32->f16 through VGPRs (coalesced float2);
// B staged with CDNA5 async global->LDS copies (ASYNCcnt);
// B-fragment loads software-pipelined across the 12 N-tiles.
// ---------------------------------------------------------------------------
__global__ __launch_bounds__(256) void gemm_kernel(
    const float* __restrict__ A, const _Float16* __restrict__ bft,
    float* __restrict__ C) {
  __shared__ _Float16 ash[2][MBLK][KSTEP + 4];   // +4 f16 pad: conflict-free ds
  __shared__ _Float16 bsh[2][NCOL][KSTEP + 4];

  const int tid   = threadIdx.x;
  const int wave  = tid >> 5;
  const int lane  = tid & 31;
  const int l16   = lane & 15;
  const int lhalf = lane >> 4;               // 0: lanes 0-15, 1: lanes 16-31
  const int m0    = blockIdx.x * MBLK + wave * 16;
  const bool active = (m0 < NHDV);
  const int rowA   = wave * 16 + l16;        // row in ash for A fragment
  const int base8  = lhalf * 8;              // A per-lane K base (ISA layout)
  const int base16 = lhalf * 8;              // B per-lane K base (ISA layout, same as A)

  // staging thread mapping for A: 2 threads per row, 16 k each
  const int rowL  = tid >> 1;
  const int halfk = (tid & 1) * 16;
  const int grow  = blockIdx.x * MBLK + rowL;
  const float* asrc = A + (size_t)grow * NVERT + halfk;

  v8f acc[12];
#pragma unroll
  for (int t = 0; t < 12; ++t)
#pragma unroll
    for (int i = 0; i < 8; ++i) acc[t][i] = 0.0f;

  // ---- B staging: async global->LDS, 1536 x 8B chunks per tile ----
  auto stageB = [&](int ks, int bufi) {
    int k0 = ks * KSTEP;
#pragma unroll
    for (int i = 0; i < 6; ++i) {
      int idx = tid + i * 256;               // 0..1535
      int col = idx >> 3;
      int ko  = (idx & 7) * 4;               // f16 offset within KSTEP
      unsigned ldsa = (unsigned)(uintptr_t)&bsh[bufi][col][ko];
      const _Float16* g = bft + (size_t)col * KPAD + k0 + ko;
      asm volatile("global_load_async_to_lds_b64 %0, %1, off"
                   :: "v"(ldsa), "v"(g) : "memory");
    }
  };

  // ---- A staging: global f32 loads into registers ----
  auto loadA = [&](int ks, float2* r) {
    int k0 = ks * KSTEP;
    if (grow < NHDV) {
      if (ks < KFULL) {
#pragma unroll
        for (int j = 0; j < 8; ++j)
          r[j] = *(const float2*)(asrc + k0 + 2 * j);  // 8B aligned
      } else {
#pragma unroll
        for (int j = 0; j < 8; ++j) {
          int kg = k0 + halfk + 2 * j;
          float x0 = (kg < NVERT) ? asrc[k0 + 2 * j] : 0.0f;      // zero tail
          float x1 = (kg + 1 < NVERT) ? asrc[k0 + 2 * j + 1] : 0.0f;
          r[j].x = x0; r[j].y = x1;
        }
      }
    }
  };
  // ---- A staging: convert f32->f16, store to LDS ----
  auto storeA = [&](int bufi, const float2* r) {
    if (grow < NHDV) {
#pragma unroll
      for (int j = 0; j < 4; ++j) {
        v4h h;
        h[0] = (_Float16)r[2 * j].x;     h[1] = (_Float16)r[2 * j].y;
        h[2] = (_Float16)r[2 * j + 1].x; h[3] = (_Float16)r[2 * j + 1].y;
        *(v4h*)(&ash[bufi][rowL][halfk + 4 * j]) = h;
      }
    }
  };

  auto loadBfrag = [&](int bufi, int t) -> v16h {
    int col = t * 16 + l16;
    const _Float16* p = &bsh[bufi][col][base16];
    v4h b0 = *(const v4h*)(p);
    v4h b1 = *(const v4h*)(p + 4);
    v4h b2 = *(const v4h*)(p + 16);
    v4h b3 = *(const v4h*)(p + 20);
    v16h bF;
#pragma unroll
    for (int e = 0; e < 4; ++e) {
      bF[e] = b0[e]; bF[4 + e] = b1[e]; bF[8 + e] = b2[e]; bF[12 + e] = b3[e];
    }
    return bF;
  };

  auto compute = [&](int bufi) {
    const _Float16* ap = &ash[bufi][rowA][base8];
    v4h a0 = *(const v4h*)(ap);
    v4h a1 = *(const v4h*)(ap + 4);
    v4h a2 = *(const v4h*)(ap + 16);
    v4h a3 = *(const v4h*)(ap + 20);
    v16h aF;
#pragma unroll
    for (int e = 0; e < 4; ++e) {
      aF[e] = a0[e]; aF[4 + e] = a1[e]; aF[8 + e] = a2[e]; aF[12 + e] = a3[e];
    }
    v16h bCur = loadBfrag(bufi, 0);
#pragma unroll
    for (int t = 0; t < 12; ++t) {
      v16h bNxt = bCur;
      if (t < 11) bNxt = loadBfrag(bufi, t + 1);   // prefetch next N-tile
      acc[t] = __builtin_amdgcn_wmma_f32_16x16x32_f16(
          false, aF, false, bCur, (short)0, acc[t], false, false);
      bCur = bNxt;
    }
  };

  // ---- prologue: stage tile 0 into buffer 0 ----
  {
    float2 r[8];
    loadA(0, r);
    storeA(0, r);
    stageB(0, 0);
  }

  for (int ks = 0; ks <= KFULL; ++ks) {
    const int cur = ks & 1;
    asm volatile("s_wait_asynccnt 0" ::: "memory");  // async B writes visible
    __syncthreads();                                 // (compiler adds dscnt wait)
    float2 r[8];
    const bool nxt = (ks < KFULL);
    if (nxt) {                      // issue next tile's global traffic early
      loadA(ks + 1, r);
      stageB(ks + 1, cur ^ 1);
    }
    if (active) compute(cur);       // overlap WMMA with the loads above
    if (nxt) storeA(cur ^ 1, r);    // convert + LDS store after compute
  }

  if (active) {
#pragma unroll
    for (int t = 0; t < 12; ++t) {
      int col = t * 16 + l16;
      // C layout: VGPR i -> M = m0 + i + 8*lhalf; contiguous 32B-aligned store
      *(v8f*)(C + (size_t)col * NHDV + m0 + lhalf * 8) = acc[t];
    }
  }
}

// ---------------------------------------------------------------------------
// Kernel 4: fused epilogue. One block per batch. Deterministic tree reduce.
// C is transposed: C[col][n], col = 3b+d -> three coalesced streams per block.
// ---------------------------------------------------------------------------
__global__ __launch_bounds__(256) void reduce_kernel(
    const float* __restrict__ C, const int* __restrict__ face_part,
    const int* __restrict__ vert_fid, const float* __restrict__ per_part,
    float* __restrict__ out) {
  int b = blockIdx.x, tid = threadIdx.x;
  const float* pp  = per_part + b * NPART;
  const float* c0p = C + (size_t)(3 * b + 0) * NHDV;
  const float* c1p = C + (size_t)(3 * b + 1) * NHDV;
  const float* c2p = C + (size_t)(3 * b + 2) * NHDV;
  float s0 = 0.f, s2 = 0.f, w1s = 0.f, p0 = 0.f, p2 = 0.f, pws = 0.f;
  for (int n = tid; n < NHDV; n += 256) {
    float c0 = c0p[n], c1 = c1p[n], c2 = c2p[n];
    float w1 = pp[face_part[vert_fid[n]]];
    float pw = (c1 < 0.0f) ? (1.0f - 100.0f * c1) : expf(-10.0f * c1);
    s0 += c0 * w1;  s2 += c2 * w1;  w1s += w1;
    p0 += c0 * pw;  p2 += c2 * pw;  pws += pw;
  }
  __shared__ float red[6][256];
  red[0][tid] = s0; red[1][tid] = s2; red[2][tid] = w1s;
  red[3][tid] = p0; red[4][tid] = p2; red[5][tid] = pws;
  __syncthreads();
  for (int s = 128; s > 0; s >>= 1) {
    if (tid < s) {
      for (int q = 0; q < 6; ++q) red[q][tid] += red[q][tid + s];
    }
    __syncthreads();
  }
  if (tid == 0) {
    float inv1 = 1.0f / red[2][0];
    float invp = 1.0f / (red[5][0] + 1e-6f);
    float dx = red[0][0] * inv1 - red[3][0] * invp;
    float dz = red[1][0] * inv1 - red[4][0] * invp;
    out[b] = sqrtf(dx * dx + dz * dz);
  }
}

// ---------------------------------------------------------------------------
extern "C" void kernel_launch(void* const* d_in, const int* in_sizes, int n_in,
                              void* d_out, int out_size, void* d_ws, size_t ws_size,
                              hipStream_t stream) {
  const float* vertices  = (const float*)d_in[0];   // (B, NV, 3)
  const float* hd_op     = (const float*)d_in[1];   // (NHD, NV)
  const int*   faces     = (const int*)d_in[2];     // (NF, 3)
  const int*   vert_fid  = (const int*)d_in[3];     // (NHD,)
  const int*   face_part = (const int*)d_in[4];     // (NF,)
  float* out = (float*)d_out;                       // (B,)

  char* ws = (char*)d_ws;
  float*    per_part = (float*)ws;                                    //  6,144 B
  _Float16* bft      = (_Float16*)(ws + 8192);                        //  2,654,208 B
  float*    Cbuf     = (float*)(ws + 8192 + (size_t)NCOL * KPAD * 2); // 15,360,000 B

  int prep_blocks = (NCOL * KPAD + 255) / 256;       // 5184
  int gemm_blocks = (NHDV + MBLK - 1) / MBLK;        // 157

  prep_kernel<<<prep_blocks, 256, 0, stream>>>(vertices, bft);
  vol_kernel<<<BATCH, 256, 0, stream>>>(vertices, faces, face_part, per_part);
  gemm_kernel<<<gemm_blocks, 256, 0, stream>>>(hd_op, bft, Cbuf);
  reduce_kernel<<<BATCH, 256, 0, stream>>>(Cbuf, face_part, vert_fid, per_part, out);
}
